// LabelPropagationCluster_24137716203991
// MI455X (gfx1250) — compile-verified, matrix operand order checked by reading, not run on hardware
//
#include <hip/hip_runtime.h>
#include <hip/hip_bf16.h>

typedef __bf16 bf16_t;
typedef __attribute__((ext_vector_type(16))) __bf16 v16bf;
typedef __attribute__((ext_vector_type(8)))  float  v8f;

#define NROWS 10000
#define DIM   768
#define NCLS  1000
#define KNN   10
#define SLOTS 64              // KNN own edges + up to 54 transpose edges
#define ALPHAC 0.99f
#define EPSV  1e-8f

// ---------------------------------------------------------------------------
// 1) Row-normalize features -> fnorm (bf16), and cast raw features -> bf16
// ---------------------------------------------------------------------------
__global__ __launch_bounds__(256) void k_rownorm(const float* __restrict__ feat,
                                                 bf16_t* __restrict__ fnorm,
                                                 bf16_t* __restrict__ featb) {
    const int row = blockIdx.x;
    const float* fr = feat + (size_t)row * DIM;
    __shared__ float red[256];
    float s = 0.f;
    for (int k = threadIdx.x; k < DIM; k += 256) { float v = fr[k]; s += v * v; }
    red[threadIdx.x] = s; __syncthreads();
    for (int o = 128; o > 0; o >>= 1) {
        if (threadIdx.x < o) red[threadIdx.x] += red[threadIdx.x + o];
        __syncthreads();
    }
    const float inv = 1.0f / (sqrtf(red[0]) + EPSV);
    for (int k = threadIdx.x; k < DIM; k += 256) {
        float v = fr[k];
        fnorm[(size_t)row * DIM + k] = (bf16_t)(v * inv);
        featb[(size_t)row * DIM + k] = (bf16_t)v;
    }
}

__global__ __launch_bounds__(256) void k_cast_w(const float* __restrict__ w,
                                                bf16_t* __restrict__ wb) {
    int i = blockIdx.x * blockDim.x + threadIdx.x;
    if (i < NCLS * DIM) wb[i] = (bf16_t)w[i];
}

// ---------------------------------------------------------------------------
// 2) Fused sims = fnorm @ fnorm^T with streaming top-K (never materializes NxN)
//    8 wave32s per block: 32-row band x 64-col chunk, bf16 WMMA, fp32 acc.
// ---------------------------------------------------------------------------
__global__ __launch_bounds__(256) void k_sims_topk(const bf16_t* __restrict__ fnorm,
                                                   float* __restrict__ tkv,
                                                   int*   __restrict__ tki) {
    __shared__ alignas(32) bf16_t Alds[32 * DIM];   // 48 KB band of A
    __shared__ float Sl[32][65];                    // padded chunk (bank-safe)
    const int tid  = threadIdx.x;
    const int wave = tid >> 5;
    const int lane = tid & 31;
    const int row0 = blockIdx.x * 32;

    // Stage A band into LDS (16B vectors), zero-fill past N.
    {
        uint4* dst = (uint4*)Alds;
        const int totalVec = 32 * DIM * 2 / 16;     // 3072
        for (int v = tid; v < totalVec; v += 256) {
            int byteOff = v * 16;
            int r = byteOff / (DIM * 2);
            int c = byteOff % (DIM * 2);
            uint4 val = make_uint4(0u, 0u, 0u, 0u);
            int gr = row0 + r;
            if (gr < NROWS)
                val = *(const uint4*)((const char*)fnorm + (size_t)gr * DIM * 2 + c);
            dst[v] = val;
        }
    }
    __syncthreads();

    // per-row top-K state (threads 0..31 own one row each)
    float tv[KNN]; int ti[KNN];
#pragma unroll
    for (int k = 0; k < KNN; ++k) { tv[k] = -__builtin_inff(); ti[k] = 0; }
    const int myrow = row0 + tid;

    const int at    = wave >> 2;   // which 16-row A tile (0..1)
    const int bt    = wave & 3;    // which 16-col B tile (0..3)
    const int lhalf = lane >> 4;   // lane group 0/1
    const int l16   = lane & 15;

    const bf16_t* Abase = Alds + (size_t)(at * 16 + l16) * DIM + lhalf * 16;

    for (int j0 = 0; j0 < NROWS; j0 += 64) {
        int jb = j0 + bt * 16 + l16;
        if (jb >= NROWS) jb = NROWS - 1;            // clamped garbage is masked below
        const bf16_t* Bbase = fnorm + (size_t)jb * DIM + lhalf * 16;

        v8f acc = {};
#pragma unroll 4
        for (int k0 = 0; k0 < DIM; k0 += 32) {
            v16bf a = *(const v16bf*)(Abase + k0);
            v16bf b = *(const v16bf*)(Bbase + k0);
            acc = __builtin_amdgcn_wmma_f32_16x16x32_bf16(
                      false, a, false, b, (short)0, acc, false, false);
        }
        // 16x16 fp32 tile -> LDS (C/D layout: VGPR r = rows r / r+8)
#pragma unroll
        for (int r = 0; r < 8; ++r) {
            int m = r + lhalf * 8;
            Sl[at * 16 + m][bt * 16 + l16] = acc[r];
        }
        __syncthreads();

        // branchless register top-10 insertion, one thread per row
        if (tid < 32 && myrow < NROWS) {
#pragma unroll 1
            for (int c = 0; c < 64; ++c) {
                int col = j0 + c;
                float v = Sl[tid][c];
                if (col < NROWS && col != myrow && v > tv[KNN - 1]) {
                    int rank = 0;
#pragma unroll
                    for (int p = 0; p < KNN - 1; ++p) rank += (tv[p] >= v) ? 1 : 0;
#pragma unroll
                    for (int p = KNN - 1; p > 0; --p)
                        if (tv[p - 1] < v) { tv[p] = tv[p - 1]; ti[p] = ti[p - 1]; }
#pragma unroll
                    for (int p = 0; p < KNN; ++p)
                        if (p == rank) { tv[p] = v; ti[p] = col; }
                }
            }
        }
        __syncthreads();
    }

    if (tid < 32 && myrow < NROWS) {
#pragma unroll
        for (int k = 0; k < KNN; ++k) {
            tkv[(size_t)myrow * KNN + k] = tv[k];
            tki[(size_t)myrow * KNN + k] = ti[k];
        }
    }
}

// ---------------------------------------------------------------------------
// 3) Sparse graph build: W = 0.5*(W + W^T), degree-normalize -> S
// ---------------------------------------------------------------------------
__global__ void k_slots_init(int* __restrict__ sc, float* __restrict__ sv,
                             int* __restrict__ cnt) {
    int i = blockIdx.x * blockDim.x + threadIdx.x;
    int stride = gridDim.x * blockDim.x;
    for (int t = i; t < NROWS * SLOTS; t += stride) { sc[t] = 0; sv[t] = 0.f; }
    for (int t = i; t < NROWS; t += stride) cnt[t] = 0;
}

__global__ void k_scatter(const float* __restrict__ tkv, const int* __restrict__ tki,
                          int* __restrict__ sc, float* __restrict__ sv,
                          int* __restrict__ cnt) {
    int t = blockIdx.x * blockDim.x + threadIdx.x;
    if (t >= NROWS * KNN) return;
    int i = t / KNN, k = t % KNN;
    float v = tkv[t];
    v = fmaxf(v, 0.f); v = v * v * v * 0.5f;     // max(v,0)^3, halved for symmetrization
    int j = tki[t];
    sc[i * SLOTS + k] = j; sv[i * SLOTS + k] = v;   // forward edge
    int pos = atomicAdd(&cnt[j], 1);                 // transpose edge
    if (pos < SLOTS - KNN) {
        sc[(size_t)j * SLOTS + KNN + pos] = i;
        sv[(size_t)j * SLOTS + KNN + pos] = v;
    }
}

__global__ void k_degree(const float* __restrict__ sv, const int* __restrict__ cnt,
                         float* __restrict__ dinv, int* __restrict__ scount) {
    int i = blockIdx.x * blockDim.x + threadIdx.x;
    if (i >= NROWS) return;
    int c = cnt[i]; if (c > SLOTS - KNN) c = SLOTS - KNN;
    int tot = KNN + c;
    float s = 0.f;
    for (int k = 0; k < tot; ++k) s += sv[(size_t)i * SLOTS + k];
    dinv[i] = rsqrtf(s + EPSV);
    scount[i] = tot;
}

__global__ void k_scale(float* __restrict__ sv, const int* __restrict__ sc,
                        const float* __restrict__ dinv, const int* __restrict__ scount) {
    int t = blockIdx.x * blockDim.x + threadIdx.x;
    if (t >= NROWS * SLOTS) return;
    int i = t / SLOTS, k = t % SLOTS;
    if (k < scount[i]) sv[t] = sv[t] * dinv[i] * dinv[sc[t]];
}

// ---------------------------------------------------------------------------
// 4) logits = features @ Wcls^T via bf16 WMMA (10000x768x1000)
// ---------------------------------------------------------------------------
__global__ __launch_bounds__(256) void k_logits(const bf16_t* __restrict__ featb,
                                                const bf16_t* __restrict__ wclsb,
                                                float* __restrict__ logits) {
    __shared__ alignas(32) bf16_t Alds[32 * DIM];
    const int tid  = threadIdx.x;
    const int wave = tid >> 5;
    const int lane = tid & 31;
    const int row0 = blockIdx.x * 32;

    {
        uint4* dst = (uint4*)Alds;
        const int totalVec = 32 * DIM * 2 / 16;
        for (int v = tid; v < totalVec; v += 256) {
            int byteOff = v * 16;
            int r = byteOff / (DIM * 2);
            int c = byteOff % (DIM * 2);
            uint4 val = make_uint4(0u, 0u, 0u, 0u);
            int gr = row0 + r;
            if (gr < NROWS)
                val = *(const uint4*)((const char*)featb + (size_t)gr * DIM * 2 + c);
            dst[v] = val;
        }
    }
    __syncthreads();

    const int at = wave >> 2, bt = wave & 3;
    const int lhalf = lane >> 4, l16 = lane & 15;
    const bf16_t* Abase = Alds + (size_t)(at * 16 + l16) * DIM + lhalf * 16;

    for (int j0 = 0; j0 < 1024; j0 += 64) {          // 16 chunks cover C=1000
        int jb = j0 + bt * 16 + l16;
        if (jb >= NCLS) jb = NCLS - 1;
        const bf16_t* Bbase = wclsb + (size_t)jb * DIM + lhalf * 16;

        v8f acc = {};
#pragma unroll 4
        for (int k0 = 0; k0 < DIM; k0 += 32) {
            v16bf a = *(const v16bf*)(Abase + k0);
            v16bf b = *(const v16bf*)(Bbase + k0);
            acc = __builtin_amdgcn_wmma_f32_16x16x32_bf16(
                      false, a, false, b, (short)0, acc, false, false);
        }
#pragma unroll
        for (int r = 0; r < 8; ++r) {
            int m   = r + lhalf * 8;
            int row = row0 + at * 16 + m;
            int col = j0 + bt * 16 + l16;
            if (row < NROWS && col < NCLS)
                logits[(size_t)row * NCLS + col] = acc[r];
        }
    }
}

// ---------------------------------------------------------------------------
// 5) Row softmax -> Y, and seed Z0 = Y in d_out
// ---------------------------------------------------------------------------
__global__ __launch_bounds__(256) void k_softmax(const float* __restrict__ logits,
                                                 float* __restrict__ Y,
                                                 float* __restrict__ Z0) {
    const int row = blockIdx.x;
    const float* lr = logits + (size_t)row * NCLS;
    __shared__ float red[256];
    float m = -__builtin_inff();
    for (int c = threadIdx.x; c < NCLS; c += 256) m = fmaxf(m, lr[c]);
    red[threadIdx.x] = m; __syncthreads();
    for (int o = 128; o > 0; o >>= 1) {
        if (threadIdx.x < o) red[threadIdx.x] = fmaxf(red[threadIdx.x], red[threadIdx.x + o]);
        __syncthreads();
    }
    m = red[0]; __syncthreads();
    float s = 0.f;
    for (int c = threadIdx.x; c < NCLS; c += 256) s += __expf(lr[c] - m);
    red[threadIdx.x] = s; __syncthreads();
    for (int o = 128; o > 0; o >>= 1) {
        if (threadIdx.x < o) red[threadIdx.x] += red[threadIdx.x + o];
        __syncthreads();
    }
    float inv = 1.0f / red[0];
    for (int c = threadIdx.x; c < NCLS; c += 256) {
        float y = __expf(lr[c] - m) * inv;
        Y[(size_t)row * NCLS + c]  = y;
        Z0[(size_t)row * NCLS + c] = y;
    }
}

// ---------------------------------------------------------------------------
// 6) One propagation step: Zout[i] = (1-a)*Y[i] + a * sum_s S_is * Zin[col_s]
//    One block per row, float4 gathers; Z stays L2-resident (40MB << 192MB)
// ---------------------------------------------------------------------------
__global__ __launch_bounds__(256) void k_propagate(const float* __restrict__ Zin,
                                                   const float* __restrict__ Y,
                                                   const int*   __restrict__ sc,
                                                   const float* __restrict__ sv,
                                                   const int*   __restrict__ scount,
                                                   float* __restrict__ Zout) {
    const int row = blockIdx.x;
    __shared__ int   lc[SLOTS];
    __shared__ float lv[SLOTS];
    __shared__ int   lcount;
    if (threadIdx.x == 0) lcount = scount[row];
    if (threadIdx.x < SLOTS) {
        lc[threadIdx.x] = sc[(size_t)row * SLOTS + threadIdx.x];
        lv[threadIdx.x] = sv[(size_t)row * SLOTS + threadIdx.x];
    }
    __syncthreads();
    const int cnt = lcount;
    const int t = threadIdx.x;                  // float4 lane: 0..249
    if (t < NCLS / 4) {
        float4 acc = make_float4(0.f, 0.f, 0.f, 0.f);
        for (int s = 0; s < cnt; ++s) {
            const float4 z = *(const float4*)(Zin + (size_t)lc[s] * NCLS + t * 4);
            const float w = lv[s];
            acc.x += w * z.x; acc.y += w * z.y; acc.z += w * z.z; acc.w += w * z.w;
        }
        const float4 y = *(const float4*)(Y + (size_t)row * NCLS + t * 4);
        float4 o;
        o.x = ALPHAC * acc.x + (1.f - ALPHAC) * y.x;
        o.y = ALPHAC * acc.y + (1.f - ALPHAC) * y.y;
        o.z = ALPHAC * acc.z + (1.f - ALPHAC) * y.z;
        o.w = ALPHAC * acc.w + (1.f - ALPHAC) * y.w;
        *(float4*)(Zout + (size_t)row * NCLS + t * 4) = o;
    }
}

// ---------------------------------------------------------------------------
extern "C" void kernel_launch(void* const* d_in, const int* in_sizes, int n_in,
                              void* d_out, int out_size, void* d_ws, size_t ws_size,
                              hipStream_t stream) {
    const float* feat = (const float*)d_in[0];
    const float* wcls = (const float*)d_in[1];
    float* Zfinal = (float*)d_out;

    char* ws = (char*)d_ws;
    size_t off = 0;
    auto alloc = [&](size_t bytes) -> char* {
        char* p = ws + off;
        off += (bytes + 255) & ~(size_t)255;
        return p;
    };
    bf16_t* fnorm  = (bf16_t*)alloc((size_t)NROWS * DIM * 2);
    bf16_t* featb  = (bf16_t*)alloc((size_t)NROWS * DIM * 2);
    bf16_t* wclsb  = (bf16_t*)alloc((size_t)NCLS  * DIM * 2);
    float*  tkv    = (float*) alloc((size_t)NROWS * KNN * 4);
    int*    tki    = (int*)   alloc((size_t)NROWS * KNN * 4);
    int*    sc     = (int*)   alloc((size_t)NROWS * SLOTS * 4);
    float*  sv     = (float*) alloc((size_t)NROWS * SLOTS * 4);
    int*    cnt    = (int*)   alloc((size_t)NROWS * 4);
    int*    scnt   = (int*)   alloc((size_t)NROWS * 4);
    float*  dinv   = (float*) alloc((size_t)NROWS * 4);
    float*  logits = (float*) alloc((size_t)NROWS * NCLS * 4);   // reused as Z ping
    float*  Y      = (float*) alloc((size_t)NROWS * NCLS * 4);
    float*  Zb     = logits;                                     // alias after softmax

    (void)in_sizes; (void)n_in; (void)out_size; (void)ws_size;

    k_cast_w   <<<(NCLS * DIM + 255) / 256, 256, 0, stream>>>(wcls, wclsb);
    k_rownorm  <<<NROWS, 256, 0, stream>>>(feat, fnorm, featb);
    k_sims_topk<<<(NROWS + 31) / 32, 256, 0, stream>>>(fnorm, tkv, tki);
    k_slots_init<<<1024, 256, 0, stream>>>(sc, sv, cnt);
    k_scatter  <<<(NROWS * KNN + 255) / 256, 256, 0, stream>>>(tkv, tki, sc, sv, cnt);
    k_degree   <<<(NROWS + 255) / 256, 256, 0, stream>>>(sv, cnt, dinv, scnt);
    k_scale    <<<(NROWS * SLOTS + 255) / 256, 256, 0, stream>>>(sv, sc, dinv, scnt);
    k_logits   <<<(NROWS + 31) / 32, 256, 0, stream>>>(featb, wclsb, logits);
    k_softmax  <<<NROWS, 256, 0, stream>>>(logits, Y, Zfinal);   // Z0 = Y in d_out

    float* zA = Zfinal;   // 20 (even) steps -> result ends in d_out
    float* zB = Zb;
    for (int it = 0; it < 20; ++it) {
        k_propagate<<<NROWS, 256, 0, stream>>>(zA, Y, sc, sv, scnt, zB);
        float* tmp = zA; zA = zB; zB = tmp;
    }
}
